// TransformerFlowModel_82755429859929
// MI455X (gfx1250) — matrix operand order repaired and checked by reading, not compiled, not run on hardware
//
#include <hip/hip_runtime.h>
#include <hip/hip_bf16.h>
#include <math.h>

// ---------------------------------------------------------------------------
// Types
// ---------------------------------------------------------------------------
typedef _Float16 h16;
typedef _Float16 v16h __attribute__((ext_vector_type(16)));
typedef _Float16 v8h  __attribute__((ext_vector_type(8)));
typedef float    v8f  __attribute__((ext_vector_type(8)));
typedef int      v4i  __attribute__((ext_vector_type(4)));

#define D_   1024
#define H_   16
#define HD_  64
#define L_   8
#define V_   2048
#define FF_  4096
#define WIN_ 256
#define B_   2
#define S_   1024
#define M_   (B_*S_)
#define CIN_ 192   // ME+RE+TE+TPE+DE

// gfx1250 async global->LDS copy (ASYNCcnt path); fall back to plain loads.
#if defined(__AMDGCN__) && __has_builtin(__builtin_amdgcn_global_load_async_to_lds_b128)
#define USE_ASYNC 1
typedef __attribute__((address_space(1))) v4i gv4i;   // global int4
typedef __attribute__((address_space(3))) v4i lv4i;   // LDS int4
#else
#define USE_ASYNC 0
#endif

// ---------------------------------------------------------------------------
// WMMA helpers (CDNA5 wave32, 16x16x32 f16 -> f32)
// ---------------------------------------------------------------------------
__device__ inline v8f wmma_f16(v16h a, v16h b, v8f c) {
  return __builtin_amdgcn_wmma_f32_16x16x32_f16(false, a, false, b,
                                                (short)0, c, false, false);
}

// A-matrix fragment (16x32, ISA 7.12.2): lane m=lane&15, g=lane>>4:
// halves 0..7 = K[8g..8g+7], 8..15 = K[16+8g..23+8g]
__device__ inline v16h frag_a(const h16* rowptr, int g) {
  v8h lo = *(const v8h*)(rowptr + 8*g);
  v8h hi = *(const v8h*)(rowptr + 16 + 8*g);
  v16h r;
#pragma unroll
  for (int i = 0; i < 8; ++i) { r[i] = lo[i]; r[8+i] = hi[i]; }
  return r;
}

// B-matrix fragment (32x16): lane n=lane&15, g=lane>>4:
// halves 0..7 = K[8g..8g+7], 8..15 = K[16+8g..23+8g]
__device__ inline v16h frag_b(const h16* colptr, int g) {
  v8h lo = *(const v8h*)(colptr + 8*g);
  v8h hi = *(const v8h*)(colptr + 16 + 8*g);
  v16h r;
#pragma unroll
  for (int i = 0; i < 8; ++i) { r[i] = lo[i]; r[8+i] = hi[i]; }
  return r;
}

// Wait for this wave's async copies, then block barrier.
__device__ inline void wait_async_and_barrier() {
#if USE_ASYNC
#if __has_builtin(__builtin_amdgcn_s_wait_asynccnt)
  __builtin_amdgcn_s_wait_asynccnt(0);
#else
  asm volatile("s_wait_asynccnt 0" ::: "memory");
#endif
#endif
  __syncthreads();
}

// Stage one 128x32-half tile into LDS: 512 x 16B chunks, 2 per thread.
__device__ inline void stage_tile(const h16* __restrict__ gbase, int ldg,
                                  h16 (* __restrict__ lds)[40], int tid)
{
#pragma unroll
  for (int q = 0; q < 2; ++q) {
    int c   = (tid << 1) | q;     // 0..511
    int row = c >> 2;             // 0..127
    int off = (c & 3) << 3;       // 0,8,16,24 halves
    const h16* src = gbase + (size_t)row * ldg + off;
#if USE_ASYNC
    __builtin_amdgcn_global_load_async_to_lds_b128(
        (gv4i*)src, (lv4i*)&lds[row][off], 0, 0);
#else
    *(v8h*)&lds[row][off] = *(const v8h*)src;
#endif
  }
}

// ---------------------------------------------------------------------------
// GEMM: C[M,N] = A[M,K] * W[N,K]^T (+resid), f16 operands, f32 accum,
// output f32 or f16. 128x128 block tile, BK=32, double-buffered async LDS.
// ---------------------------------------------------------------------------
template<int RESID, int OUT16>
__global__ __launch_bounds__(256)
void gemm_nt16(const h16* __restrict__ A, const h16* __restrict__ W,
               const float* __restrict__ Rres, float* __restrict__ Cf,
               h16* __restrict__ Ch, int M, int N, int K)
{
  __shared__ __align__(16) h16 As[2][128][40];
  __shared__ __align__(16) h16 Ws[2][128][40];

  const int tid  = threadIdx.x;
  const int wid  = tid >> 5;
  const int lane = tid & 31;
  const int g    = lane >> 4;
  const int ln16 = lane & 15;
  const int wm   = wid >> 2;   // 0..1 -> 64-row slab
  const int wn   = wid & 3;    // 0..3 -> 32-col slab
  const int bm   = blockIdx.y * 128;
  const int bn   = blockIdx.x * 128;

  v8f acc[4][2];
#pragma unroll
  for (int i = 0; i < 4; ++i)
#pragma unroll
    for (int j = 0; j < 2; ++j)
#pragma unroll
      for (int e = 0; e < 8; ++e) acc[i][j][e] = 0.0f;

  const h16* Ab = A + (size_t)bm * K;
  const h16* Wb = W + (size_t)bn * K;

  stage_tile(Ab, K, As[0], tid);
  stage_tile(Wb, K, Ws[0], tid);
  wait_async_and_barrier();

  const int nk = K >> 5;
  for (int t = 0; t < nk; ++t) {
    const int cur = t & 1;
    if (t + 1 < nk) {
      stage_tile(Ab + (size_t)(t + 1) * 32, K, As[cur ^ 1], tid);
      stage_tile(Wb + (size_t)(t + 1) * 32, K, Ws[cur ^ 1], tid);
    }
    v16h bfr[2];
#pragma unroll
    for (int j = 0; j < 2; ++j)
      bfr[j] = frag_b(&Ws[cur][wn*32 + j*16 + ln16][0], g);
#pragma unroll
    for (int i = 0; i < 4; ++i) {
      v16h afr = frag_a(&As[cur][wm*64 + i*16 + ln16][0], g);
#pragma unroll
      for (int j = 0; j < 2; ++j)
        acc[i][j] = wmma_f16(afr, bfr[j], acc[i][j]);
    }
    wait_async_and_barrier();
  }

#pragma unroll
  for (int i = 0; i < 4; ++i)
#pragma unroll
    for (int j = 0; j < 2; ++j)
#pragma unroll
      for (int v = 0; v < 8; ++v) {
        int row = bm + wm*64 + i*16 + v + 8*g;
        int col = bn + wn*32 + j*16 + ln16;
        float val = acc[i][j][v];
        if (RESID) val += Rres[(size_t)row * N + col];
        if (OUT16) Ch[(size_t)row * N + col] = (h16)val;
        else       Cf[(size_t)row * N + col] = val;
      }
}

// ---------------------------------------------------------------------------
// Windowed causal flash attention with WMMA; f16 output for o-projection.
// grid: (S/64, H, B); block 256 = 8 waves; wave = 16-query tile, full head dim
// ---------------------------------------------------------------------------
__global__ __launch_bounds__(256)
void attn_win(const float* __restrict__ Qg, const float* __restrict__ Kg,
              const float* __restrict__ Vg, h16* __restrict__ Y16)
{
  __shared__ __align__(16) h16 Ks[32][72];      // [key][d]
  __shared__ __align__(16) h16 Vt[64][40];      // [d][key] (transposed)
  __shared__ __align__(16) h16 Ps[4][16][32];   // per-wave P staging

  const int tid  = threadIdx.x;
  const int wid  = tid >> 5;
  const int lane = tid & 31;
  const int g    = lane >> 4;
  const int ln16 = lane & 15;

  const int qb   = blockIdx.x * 64;
  const int head = blockIdx.y;
  const int b    = blockIdx.z;
  const int qw   = qb + wid * 16;

  v16h qf[2];
  {
    const float* qrow = Qg + (size_t)(b*S_ + qw + ln16) * D_ + head*HD_;
    const float sc = 0.125f;   // 1/sqrt(HD)
#pragma unroll
    for (int ks = 0; ks < 2; ++ks) {
      float4 a0 = *(const float4*)(qrow + ks*32 + 8*g);
      float4 a1 = *(const float4*)(qrow + ks*32 + 8*g + 4);
      float4 b0 = *(const float4*)(qrow + ks*32 + 16 + 8*g);
      float4 b1 = *(const float4*)(qrow + ks*32 + 16 + 8*g + 4);
      v16h f;
      f[0]=(h16)(a0.x*sc); f[1]=(h16)(a0.y*sc); f[2]=(h16)(a0.z*sc); f[3]=(h16)(a0.w*sc);
      f[4]=(h16)(a1.x*sc); f[5]=(h16)(a1.y*sc); f[6]=(h16)(a1.z*sc); f[7]=(h16)(a1.w*sc);
      f[8]=(h16)(b0.x*sc); f[9]=(h16)(b0.y*sc); f[10]=(h16)(b0.z*sc); f[11]=(h16)(b0.w*sc);
      f[12]=(h16)(b1.x*sc); f[13]=(h16)(b1.y*sc); f[14]=(h16)(b1.z*sc); f[15]=(h16)(b1.w*sc);
      qf[ks] = f;
    }
  }

  float mrow[8], lrow[8];
  v8f O[4];
#pragma unroll
  for (int v = 0; v < 8; ++v) { mrow[v] = -INFINITY; lrow[v] = 0.0f; }
#pragma unroll
  for (int t = 0; t < 4; ++t)
#pragma unroll
    for (int v = 0; v < 8; ++v) O[t][v] = 0.0f;

  int ks0 = qb - (WIN_ - 1);
  if (ks0 < 0) ks0 = 0;
  ks0 &= ~31;

  for (int kc = ks0; kc < qb + 64; kc += 32) {
    __syncthreads();
    { // cooperative K/V chunk staging (f32 -> f16), V transposed
      int r  = tid >> 3;
      int d0 = (tid & 7) * 8;
      const float* kp = Kg + (size_t)(b*S_ + kc + r) * D_ + head*HD_ + d0;
      float4 k0 = *(const float4*)(kp);
      float4 k1 = *(const float4*)(kp + 4);
      Ks[r][d0+0]=(h16)k0.x; Ks[r][d0+1]=(h16)k0.y; Ks[r][d0+2]=(h16)k0.z; Ks[r][d0+3]=(h16)k0.w;
      Ks[r][d0+4]=(h16)k1.x; Ks[r][d0+5]=(h16)k1.y; Ks[r][d0+6]=(h16)k1.z; Ks[r][d0+7]=(h16)k1.w;
      const float* vp = Vg + (size_t)(b*S_ + kc + r) * D_ + head*HD_ + d0;
      float4 v0 = *(const float4*)(vp);
      float4 v1 = *(const float4*)(vp + 4);
      Vt[d0+0][r]=(h16)v0.x; Vt[d0+1][r]=(h16)v0.y; Vt[d0+2][r]=(h16)v0.z; Vt[d0+3][r]=(h16)v0.w;
      Vt[d0+4][r]=(h16)v1.x; Vt[d0+5][r]=(h16)v1.y; Vt[d0+6][r]=(h16)v1.z; Vt[d0+7][r]=(h16)v1.w;
    }
    __syncthreads();

    v8f sfr[2];
#pragma unroll
    for (int t = 0; t < 2; ++t) {
#pragma unroll
      for (int e = 0; e < 8; ++e) sfr[t][e] = 0.0f;
#pragma unroll
      for (int ks = 0; ks < 2; ++ks) {
        v16h bf = frag_b(&Ks[t*16 + ln16][ks*32], g);
        sfr[t] = wmma_f16(qf[ks], bf, sfr[t]);
      }
    }

    float pv[2][8];
    float rowmax[8];
#pragma unroll
    for (int v = 0; v < 8; ++v) {
      int qrow = qw + v + 8*g;
      float best = -INFINITY;
#pragma unroll
      for (int t = 0; t < 2; ++t) {
        int key = kc + t*16 + ln16;
        bool ok = (key <= qrow) && (qrow - key < WIN_);
        float s = ok ? sfr[t][v] : -INFINITY;
        pv[t][v] = s;
        best = fmaxf(best, s);
      }
      for (int off = 8; off; off >>= 1)
        best = fmaxf(best, __shfl_xor(best, off, 32));
      rowmax[v] = best;
    }

    float factor[8];
#pragma unroll
    for (int v = 0; v < 8; ++v) {
      float mo = mrow[v];
      float mn = fmaxf(mo, rowmax[v]);
      float f;
      if (mn == -INFINITY)       f = 1.0f;
      else if (mo == -INFINITY)  f = 0.0f;
      else                       f = __expf(mo - mn);
      mrow[v]   = mn;
      factor[v] = f;
      float psum = 0.0f;
#pragma unroll
      for (int t = 0; t < 2; ++t) {
        float s = pv[t][v];
        float p = (s == -INFINITY) ? 0.0f : __expf(s - mn);
        pv[t][v] = p;
        psum += p;
      }
      for (int off = 8; off; off >>= 1)
        psum += __shfl_xor(psum, off, 32);
      lrow[v] = lrow[v] * f + psum;
    }

#pragma unroll
    for (int v = 0; v < 8; ++v) {
      Ps[wid][v + 8*g][ln16]      = (h16)pv[0][v];
      Ps[wid][v + 8*g][16 + ln16] = (h16)pv[1][v];
    }
    __syncthreads();

#pragma unroll
    for (int t = 0; t < 4; ++t)
#pragma unroll
      for (int v = 0; v < 8; ++v) O[t][v] *= factor[v];

    v16h pf = frag_a(&Ps[wid][ln16][0], g);
#pragma unroll
    for (int t = 0; t < 4; ++t) {
      v16h vf = frag_b(&Vt[t*16 + ln16][0], g);
      O[t] = wmma_f16(pf, vf, O[t]);
    }
  }

#pragma unroll
  for (int t = 0; t < 4; ++t)
#pragma unroll
    for (int v = 0; v < 8; ++v) {
      int qrow = qw + v + 8*g;
      float inv = 1.0f / lrow[v];
      Y16[(size_t)(b*S_ + qrow) * D_ + head*HD_ + t*16 + ln16] = (h16)(O[t][v] * inv);
    }
}

// ---------------------------------------------------------------------------
// RMSNorm with f16 output (feeds GEMMs)
// ---------------------------------------------------------------------------
__global__ __launch_bounds__(256)
void rmsnorm16_k(const float* __restrict__ X, const float* __restrict__ Wt,
                 h16* __restrict__ Hout)
{
  __shared__ float red[256];
  int row = blockIdx.x;
  const float* x = X + (size_t)row * D_;
  float ss = 0.0f;
  for (int i = threadIdx.x; i < D_; i += 256) { float v = x[i]; ss += v * v; }
  red[threadIdx.x] = ss; __syncthreads();
  for (int s = 128; s; s >>= 1) {
    if (threadIdx.x < s) red[threadIdx.x] += red[threadIdx.x + s];
    __syncthreads();
  }
  float r = rsqrtf(red[0] / (float)D_ + 1e-5f);
  for (int i = threadIdx.x; i < D_; i += 256)
    Hout[(size_t)row * D_ + i] = (h16)(Wt[i] * x[i] * r);
}

// ---------------------------------------------------------------------------
// f32 -> f16 conversion (weight streaming), n multiple of 4
// ---------------------------------------------------------------------------
__global__ void cvt16_k(const float* __restrict__ src, h16* __restrict__ dst, int n)
{
  int i = (blockIdx.x * blockDim.x + threadIdx.x) * 4;
  if (i < n) {
    float4 v = *(const float4*)(src + i);
    dst[i+0] = (h16)v.x; dst[i+1] = (h16)v.y;
    dst[i+2] = (h16)v.z; dst[i+3] = (h16)v.w;
  }
}

// ---------------------------------------------------------------------------
// RoPE in-place on f32 q,k: layout [b, s, h, hd]
// ---------------------------------------------------------------------------
__global__ void rope_k(float* __restrict__ q, float* __restrict__ k)
{
  int idx = blockIdx.x * blockDim.x + threadIdx.x;   // B*S*H*32 = 2^20
  int t  = idx & 31;
  int hh = (idx >> 5) & (H_ - 1);
  int sq = (idx >> 9) & (S_ - 1);
  int b  = idx >> 19;
  int j0 = 2 * t, j1 = 2 * t + 1;
  const float lnth = 9.210340371976184f;   // ln(10000)
  float f0 = __expf(-lnth * (float)(j0 & 31) / 32.0f);
  float f1 = __expf(-lnth * (float)(j1 & 31) / 32.0f);
  float a0 = (float)sq * f0, a1 = (float)sq * f1;
  float c0 = cosf(a0), s0 = sinf(a0), c1 = cosf(a1), s1 = sinf(a1);
  size_t base = (size_t)(b*S_ + sq) * D_ + hh * HD_;
  float x0 = q[base + j0], x1 = q[base + j1];
  q[base + j0] = x0 * c0 - x1 * s0;
  q[base + j1] = x1 * c1 + x0 * s1;
  x0 = k[base + j0]; x1 = k[base + j1];
  k[base + j0] = x0 * c0 - x1 * s0;
  k[base + j1] = x1 * c1 + x0 * s1;
}

// ---------------------------------------------------------------------------
// SwiGLU on f16: g1 = silu(g1) * g3
// ---------------------------------------------------------------------------
__global__ void swiglu16_k(h16* __restrict__ g1, const h16* __restrict__ g3, int n)
{
  int i = blockIdx.x * blockDim.x + threadIdx.x;
  if (i < n) {
    float a = (float)g1[i];
    float sig = 1.0f / (1.0f + __expf(-a));
    g1[i] = (h16)(a * sig * (float)g3[i]);
  }
}

// ---------------------------------------------------------------------------
// Embedding assembly (f32 residual stream)
// ---------------------------------------------------------------------------
__global__ void embed_k(const int* __restrict__ tokens, const float* __restrict__ tok_emb,
                        const float* __restrict__ cond, const float* __restrict__ cycle_emb,
                        const float* __restrict__ strength_emb, const int* __restrict__ cl_ptr,
                        float* __restrict__ X)
{
  int i = blockIdx.x * blockDim.x + threadIdx.x;   // B*S*D = 2^21
  int dd = i & (D_ - 1);
  int s  = (i >> 10) & (S_ - 1);
  int b  = i >> 20;
  int cl = *cl_ptr;
  int cyc = cl < 16 ? cl : 16;
  int tok = tokens[b*S_ + s];
  int strength = (s % cl == 0) ? 0 : 3;
  X[i] = tok_emb[(size_t)tok * D_ + dd] + cond[b*D_ + dd]
       + cycle_emb[(s % cyc) * D_ + dd] + strength_emb[strength * D_ + dd];
}

// ---------------------------------------------------------------------------
// Conditioning: concat -> proj -> exact gelu -> layernorm.  One block per b.
// ---------------------------------------------------------------------------
__global__ __launch_bounds__(256)
void cond_k(const int* mood, const int* raga, const int* taal,
            const float* tempo, const float* duration,
            const float* mood_emb, const float* raga_emb, const float* taal_emb,
            const float* tempo_w, const float* tempo_b,
            const float* dur_w, const float* dur_b,
            const float* proj_w, const float* proj_b,
            const float* ln_g, const float* ln_b,
            float* __restrict__ cond)
{
  __shared__ float pre[CIN_];
  __shared__ float out[D_];
  __shared__ float red[256];
  int b = blockIdx.x, tid = threadIdx.x;
  if (tid < CIN_) {
    float v;
    if (tid < 64)        v = mood_emb[mood[b]*64 + tid];
    else if (tid < 128)  v = raga_emb[raga[b]*64 + (tid - 64)];
    else if (tid < 160)  v = taal_emb[taal[b]*32 + (tid - 128)];
    else if (tid < 176)  v = tempo[b]*tempo_w[tid-160] + tempo_b[tid-160];
    else                 v = duration[b]*dur_w[tid-176] + dur_b[tid-176];
    pre[tid] = v;
  }
  __syncthreads();
  for (int d = tid; d < D_; d += 256) {
    float acc = proj_b[d];
    for (int c = 0; c < CIN_; ++c) acc += pre[c] * proj_w[(size_t)d*CIN_ + c];
    out[d] = 0.5f * acc * (1.0f + erff(acc * 0.7071067811865475f));
  }
  __syncthreads();
  float sum = 0.0f;
  for (int d = tid; d < D_; d += 256) sum += out[d];
  red[tid] = sum; __syncthreads();
  for (int s = 128; s; s >>= 1) { if (tid < s) red[tid] += red[tid+s]; __syncthreads(); }
  float mu = red[0] / (float)D_;
  __syncthreads();
  float vs = 0.0f;
  for (int d = tid; d < D_; d += 256) { float dv = out[d] - mu; vs += dv * dv; }
  red[tid] = vs; __syncthreads();
  for (int s = 128; s; s >>= 1) { if (tid < s) red[tid] += red[tid+s]; __syncthreads(); }
  float rstd = rsqrtf(red[0] / (float)D_ + 1e-5f);
  for (int d = tid; d < D_; d += 256)
    cond[b*D_ + d] = (out[d] - mu) * rstd * ln_g[d] + ln_b[d];
}

// ---------------------------------------------------------------------------
// Launch
// ---------------------------------------------------------------------------
extern "C" void kernel_launch(void* const* d_in, const int* in_sizes, int n_in,
                              void* d_out, int out_size, void* d_ws, size_t ws_size,
                              hipStream_t stream)
{
  (void)in_sizes; (void)n_in; (void)out_size; (void)ws_size;
  const int*   tokens    = (const int*)  d_in[0];
  const int*   mood      = (const int*)  d_in[1];
  const int*   raga      = (const int*)  d_in[2];
  const int*   taal      = (const int*)  d_in[3];
  const float* tempo     = (const float*)d_in[4];
  const float* duration  = (const float*)d_in[5];
  const int*   cyclen    = (const int*)  d_in[6];
  const float* tok_emb   = (const float*)d_in[7];
  const float* mood_emb  = (const float*)d_in[8];
  const float* raga_emb  = (const float*)d_in[9];
  const float* taal_emb  = (const float*)d_in[10];
  const float* tempo_w   = (const float*)d_in[11];
  const float* tempo_b   = (const float*)d_in[12];
  const float* dur_w     = (const float*)d_in[13];
  const float* dur_b     = (const float*)d_in[14];
  const float* proj_w    = (const float*)d_in[15];
  const float* proj_b    = (const float*)d_in[16];
  const float* ln_g      = (const float*)d_in[17];
  const float* ln_b      = (const float*)d_in[18];
  const float* cycle_emb = (const float*)d_in[19];
  const float* str_emb   = (const float*)d_in[20];
  const float* l_anorm   = (const float*)d_in[21];
  const float* l_fnorm   = (const float*)d_in[22];
  const float* l_q       = (const float*)d_in[23];
  const float* l_k       = (const float*)d_in[24];
  const float* l_v       = (const float*)d_in[25];
  const float* l_o       = (const float*)d_in[26];
  const float* l_w1      = (const float*)d_in[27];
  const float* l_w2      = (const float*)d_in[28];
  const float* l_w3      = (const float*)d_in[29];
  const float* fnorm     = (const float*)d_in[30];
  const float* head_w    = (const float*)d_in[31];
  float* out = (float*)d_out;

  // ---- workspace carve-up -------------------------------------------------
  float* ws = (float*)d_ws;
  size_t o = 0;
  float* x    = ws + o;  o += (size_t)M_ * D_;     // residual stream (f32)
  float* qb   = ws + o;  o += (size_t)M_ * D_;
  float* kb   = ws + o;  o += (size_t)M_ * D_;
  float* vb   = ws + o;  o += (size_t)M_ * D_;
  float* cond = ws + o;  o += (size_t)B_ * D_ + 1024;  // keep next region 16B-aligned

  h16* f16 = (h16*)(ws + o);
  size_t ho = 0;
  h16* h16buf = f16 + ho;  ho += (size_t)M_ * D_;      // rmsnorm out
  h16* y16    = f16 + ho;  ho += (size_t)M_ * D_;      // attention out
  h16* g1h    = f16 + ho;  ho += (size_t)M_ * FF_;     // w1 out / swiglu out
  h16* g3h    = f16 + ho;  ho += (size_t)M_ * FF_;     // w3 out
  h16* warena = f16 + ho;                              // weight f16 arena (16M halves)
  h16* wq16 = warena;
  h16* wk16 = warena + (size_t)1*D_*D_;
  h16* wv16 = warena + (size_t)2*D_*D_;
  h16* wo16 = warena + (size_t)3*D_*D_;
  h16* w116 = warena + (size_t)4*D_*D_;
  h16* w316 = w116 + (size_t)FF_*D_;
  h16* w216 = w316 + (size_t)FF_*D_;

  cond_k<<<B_, 256, 0, stream>>>(mood, raga, taal, tempo, duration,
                                 mood_emb, raga_emb, taal_emb,
                                 tempo_w, tempo_b, dur_w, dur_b,
                                 proj_w, proj_b, ln_g, ln_b, cond);
  embed_k<<<(B_*S_*D_)/256, 256, 0, stream>>>(tokens, tok_emb, cond,
                                              cycle_emb, str_emb, cyclen, x);

  dim3 gD(D_/128, M_/128), gFF(FF_/128, M_/128), gV(V_/128, M_/128);
  const int nDD = D_*D_, nFD = FF_*D_;
  for (int l = 0; l < L_; ++l) {
    // stream this layer's weights to f16 once
    cvt16_k<<<nDD/1024, 256, 0, stream>>>(l_q  + (size_t)l*nDD, wq16, nDD);
    cvt16_k<<<nDD/1024, 256, 0, stream>>>(l_k  + (size_t)l*nDD, wk16, nDD);
    cvt16_k<<<nDD/1024, 256, 0, stream>>>(l_v  + (size_t)l*nDD, wv16, nDD);
    cvt16_k<<<nDD/1024, 256, 0, stream>>>(l_o  + (size_t)l*nDD, wo16, nDD);
    cvt16_k<<<nFD/1024, 256, 0, stream>>>(l_w1 + (size_t)l*nFD, w116, nFD);
    cvt16_k<<<nFD/1024, 256, 0, stream>>>(l_w3 + (size_t)l*nFD, w316, nFD);
    cvt16_k<<<nFD/1024, 256, 0, stream>>>(l_w2 + (size_t)l*nFD, w216, nFD);

    rmsnorm16_k<<<M_, 256, 0, stream>>>(x, l_anorm + (size_t)l*D_, h16buf);
    gemm_nt16<0,0><<<gD, 256, 0, stream>>>(h16buf, wq16, nullptr, qb, nullptr, M_, D_, D_);
    gemm_nt16<0,0><<<gD, 256, 0, stream>>>(h16buf, wk16, nullptr, kb, nullptr, M_, D_, D_);
    gemm_nt16<0,0><<<gD, 256, 0, stream>>>(h16buf, wv16, nullptr, vb, nullptr, M_, D_, D_);
    rope_k<<<(B_*S_*H_*(HD_/2))/256, 256, 0, stream>>>(qb, kb);
    attn_win<<<dim3(S_/64, H_, B_), 256, 0, stream>>>(qb, kb, vb, y16);
    gemm_nt16<1,0><<<gD, 256, 0, stream>>>(y16, wo16, x, x, nullptr, M_, D_, D_);
    rmsnorm16_k<<<M_, 256, 0, stream>>>(x, l_fnorm + (size_t)l*D_, h16buf);
    gemm_nt16<0,1><<<gFF, 256, 0, stream>>>(h16buf, w116, nullptr, nullptr, g1h, M_, FF_, D_);
    gemm_nt16<0,1><<<gFF, 256, 0, stream>>>(h16buf, w316, nullptr, nullptr, g3h, M_, FF_, D_);
    swiglu16_k<<<(M_*FF_)/256, 256, 0, stream>>>(g1h, g3h, M_*FF_);
    gemm_nt16<1,0><<<gD, 256, 0, stream>>>(g1h, w216, x, x, nullptr, M_, D_, FF_);
  }
  rmsnorm16_k<<<M_, 256, 0, stream>>>(x, fnorm, h16buf);
  cvt16_k<<<(V_*D_)/1024, 256, 0, stream>>>(head_w, warena, V_*D_);
  gemm_nt16<0,0><<<gV, 256, 0, stream>>>(h16buf, warena, nullptr, out, nullptr, M_, V_, D_);
}
